// Transition_2027224564268
// MI455X (gfx1250) — compile-verified, matrix operand order checked by reading, not run on hardware
//
#include <hip/hip_runtime.h>
#include <hip/hip_bf16.h>

// ---- problem constants (from reference) ----
#define BSZ   64
#define SIZE_ 2048
#define LEN   16
#define AD    64      // A
#define SED_  64
#define HID_  256
#define RR    65536
#define TWOA  128     // 2*A
#define XCOLS (SIZE_ + LEN)

typedef __attribute__((ext_vector_type(16))) _Float16 v16h;
typedef __attribute__((ext_vector_type(8)))  _Float16 v8h;
typedef __attribute__((ext_vector_type(8)))  float    v8f;

// ---------------------------------------------------------------------------
// WMMA helpers (CDNA5 wave32, 16x16x32 f16 -> f32)
// ---------------------------------------------------------------------------
__device__ __forceinline__ v8f wmma16(v16h a, v16h b, v8f c) {
    // 8 args: (neg_a, A, neg_b, B, c_mod, C, reuse_a, reuse_b)
    return __builtin_amdgcn_wmma_f32_16x16x32_f16(
        false, a, false, b, (short)0, c, false, false);
}

// Fragment element e holds K = (e>>3)*16 + koff + (e&7), koff = (lane>=16)*8.
// Works for A (row-major MxK) and for B when B is stored transposed (N x K,
// K contiguous).  Two aligned 16B loads per lane.
__device__ __forceinline__ v16h frag_from_row(const _Float16* rowPtr, int kBase) {
    const int lane = threadIdx.x & 31;
    const int koff = (lane >> 4) << 3;
    const _Float16* p = rowPtr + kBase + koff;
    v8h lo = *(const v8h*)(p);
    v8h hi = *(const v8h*)(p + 16);
    v16h f;
#pragma unroll
    for (int i = 0; i < 8; ++i) { f[i] = lo[i]; f[i + 8] = hi[i]; }
    return f;
}

__device__ __forceinline__ v16h frag_load(const _Float16* base, int row16,
                                          int ld, int kBase) {
    const int lane = threadIdx.x & 31;
    return frag_from_row(base + (long)(row16 + (lane & 15)) * ld, kBase);
}

// ---------------------------------------------------------------------------
// Prep kernels
// ---------------------------------------------------------------------------
__global__ void k_zero(float* p, int n) {
    int i = blockIdx.x * blockDim.x + threadIdx.x;
    if (i < n) p[i] = 0.0f;
}

__global__ void k_init_state(const int* x, float* initF) {
    int i = blockIdx.x * blockDim.x + threadIdx.x;        // B*SIZE
    if (i < BSZ * SIZE_) {
        int b = i / SIZE_, s = i % SIZE_;
        initF[i] = (float)x[b * XCOLS + s];
    }
}

__global__ void k_sentence(const int* x, const float* action_emb,
                           const float* pos_emb, float* sent) {
    int i = blockIdx.x * blockDim.x + threadIdx.x;        // B*L*2A
    if (i < BSZ * LEN * TWOA) {
        int d = i % TWOA;
        int l = (i / TWOA) % LEN;
        int b = i / (TWOA * LEN);
        float v;
        if (d < AD) {
            int a = x[b * XCOLS + SIZE_ + l];
            v = action_emb[a * AD + d];
        } else {
            v = pos_emb[l * AD + (d - AD)];
        }
        sent[i] = v;
    }
}

__global__ void k_f2h(const float* src, _Float16* dst, int n) {
    int i = blockIdx.x * blockDim.x + threadIdx.x;
    if (i < n) dst[i] = (_Float16)src[i];
}

// src is K x N row-major; dst is N x K row-major fp16 (so B-frags read K contig)
__global__ void k_transpose_f2h(const float* src, _Float16* dst, int K, int N) {
    int i = blockIdx.x * blockDim.x + threadIdx.x;
    if (i < K * N) {
        int n = i / K, k = i % K;
        dst[n * K + k] = (_Float16)src[k * N + n];
    }
}

__global__ void k_meta_init(const float* meta_init, float* meta) {
    int i = blockIdx.x * blockDim.x + threadIdx.x;        // B*A
    if (i < BSZ * AD) meta[i] = meta_init[i & (AD - 1)];
}

// ---------------------------------------------------------------------------
// relation = relu(gather(state_emb) @ w1 + b1) @ w2 + b2   -> fp16 (R x 64)
// One 16-row tile per wave; hidden (16x256) staged in LDS per wave.
// ---------------------------------------------------------------------------
__global__ void __launch_bounds__(256)
relation_kernel(const _Float16* __restrict__ sembH,
                const _Float16* __restrict__ w1t,   // 256 x 128 (N x K)
                const _Float16* __restrict__ w2t,   // 64  x 256 (N x K)
                const float* __restrict__ b1, const float* __restrict__ b2,
                const int* __restrict__ rel_subj, const int* __restrict__ rel_obj,
                _Float16* __restrict__ relH) {
    __shared__ _Float16 hidLds[8][16 * HID_];       // 8 waves * 8KB
    const int wave = threadIdx.x >> 5;
    const int lane = threadIdx.x & 31;
    const int tile = blockIdx.x * 8 + wave;          // R/16 = 4096 tiles
    const int rBase = tile * 16;
    const int rMine = rBase + (lane & 15);

    const _Float16* rowS = sembH + (long)rel_subj[rMine] * SED_;
    const _Float16* rowO = sembH + (long)rel_obj[rMine] * SED_;
    // rel_in row = [state_emb[subj] (64) | state_emb[obj] (64)], K = 128
    v16h a0 = frag_from_row(rowS, 0);
    v16h a1 = frag_from_row(rowS, 32);
    v16h a2 = frag_from_row(rowO, 0);
    v16h a3 = frag_from_row(rowO, 32);

    _Float16* hb = &hidLds[wave][0];
    const int n = lane & 15;
    const int mHi = (lane >> 4) << 3;

    // hidden = relu(rel_in @ w1 + b1): 16 N-tiles of 16 cols, K = 128
#pragma unroll 4
    for (int j = 0; j < HID_ / 16; ++j) {
        v8f acc = {};
        acc = wmma16(a0, frag_load(w1t, j * 16, TWOA, 0),  acc);
        acc = wmma16(a1, frag_load(w1t, j * 16, TWOA, 32), acc);
        acc = wmma16(a2, frag_load(w1t, j * 16, TWOA, 64), acc);
        acc = wmma16(a3, frag_load(w1t, j * 16, TWOA, 96), acc);
        float bias = b1[j * 16 + n];
#pragma unroll
        for (int e = 0; e < 8; ++e) {
            float v = acc[e] + bias;
            hb[(mHi + e) * HID_ + j * 16 + n] = (_Float16)(v > 0.0f ? v : 0.0f);
        }
    }

    // relation = hidden @ w2 + b2: 4 N-tiles of 16 cols, K = 256
#pragma unroll
    for (int j2 = 0; j2 < AD / 16; ++j2) {
        v8f acc = {};
#pragma unroll
        for (int kb = 0; kb < HID_ / 32; ++kb)
            acc = wmma16(frag_load(hb, 0, HID_, kb * 32),
                         frag_load(w2t, j2 * 16, HID_, kb * 32), acc);
        float bias = b2[j2 * 16 + n];
#pragma unroll
        for (int e = 0; e < 8; ++e)
            relH[(long)(rBase + mHi + e) * AD + j2 * 16 + n] =
                (_Float16)(acc[e] + bias);
    }
}

// ---------------------------------------------------------------------------
// Tiny per-step meta update (attention + 2-layer MLP).  One block per batch.
// ---------------------------------------------------------------------------
__global__ void __launch_bounds__(256)
meta_step_kernel(float* __restrict__ meta, _Float16* __restrict__ metaH,
                 const float* __restrict__ sent,
                 const float* __restrict__ q_w, const float* __restrict__ q_b,
                 const float* __restrict__ ms_w1, const float* __restrict__ ms_b1,
                 const float* __restrict__ ms_w2, const float* __restrict__ ms_b2) {
    const int b = blockIdx.x, tid = threadIdx.x;
    __shared__ float sm[AD], sq[TWOA], sl[LEN], sa[TWOA], sh[HID_];
    __shared__ float ssent[LEN * TWOA];

    if (tid < AD) sm[tid] = meta[b * AD + tid];
    for (int i = tid; i < LEN * TWOA; i += 256)
        ssent[i] = sent[b * LEN * TWOA + i];
    __syncthreads();

    if (tid < TWOA) {                                   // query
        float acc = q_b[tid];
        for (int k = 0; k < AD; ++k) acc += sm[k] * q_w[k * TWOA + tid];
        sq[tid] = acc;
    }
    __syncthreads();

    if (tid < LEN) {                                    // logits
        float acc = 0.0f;
        for (int d = 0; d < TWOA; ++d) acc += sq[d] * ssent[tid * TWOA + d];
        sl[tid] = acc;
    }
    __syncthreads();

    if (tid == 0) {                                     // softmax (16 values)
        float mx = sl[0];
        for (int l = 1; l < LEN; ++l) mx = fmaxf(mx, sl[l]);
        float s = 0.0f, e[LEN];
        for (int l = 0; l < LEN; ++l) { e[l] = __expf(sl[l] - mx); s += e[l]; }
        float inv = 1.0f / s;
        for (int l = 0; l < LEN; ++l) sl[l] = e[l] * inv;
    }
    __syncthreads();

    if (tid < TWOA) {                                   // attended
        float acc = 0.0f;
        for (int l = 0; l < LEN; ++l) acc += sl[l] * ssent[l * TWOA + tid];
        sa[tid] = acc;
    }
    __syncthreads();

    {                                                   // hidden (256)
        float acc = ms_b1[tid];
        for (int k = 0; k < AD; ++k)   acc += sm[k] * ms_w1[k * HID_ + tid];
        for (int k = 0; k < TWOA; ++k) acc += sa[k] * ms_w1[(AD + k) * HID_ + tid];
        sh[tid] = fmaxf(acc, 0.0f);
    }
    __syncthreads();

    if (tid < AD) {                                     // new meta (64)
        float acc = ms_b2[tid];
        for (int j = 0; j < HID_; ++j) acc += sh[j] * ms_w2[j * AD + tid];
        meta[b * AD + tid] = acc;
        metaH[b * AD + tid] = (_Float16)acc;
    }
}

// ---------------------------------------------------------------------------
// h = sigmoid(meta @ relation^T); vals = state_prev[:,subj]*h;
// nxt[:,obj] += vals   — fused WMMA GEMM + gather + scatter-add.
// One wave per 16-relation tile; wave loops over all 4 batch tiles so the
// B-fragments (relation rows) and subj/obj indices are loaded once.
// ---------------------------------------------------------------------------
__global__ void __launch_bounds__(256)
h_step_kernel(const _Float16* __restrict__ metaH,     // 64 x 64
              const _Float16* __restrict__ relH,      // R x 64 (N x K)
              const int* __restrict__ rel_subj, const int* __restrict__ rel_obj,
              const float* __restrict__ prev, int prevStride,
              float* __restrict__ nxt, int nxtStride) {
    const int nBlock = (blockIdx.x * blockDim.x + threadIdx.x) >> 5;  // R/16 waves
    const int lane = threadIdx.x & 31;

    v16h b0 = frag_load(relH, nBlock * 16, AD, 0);
    v16h b1 = frag_load(relH, nBlock * 16, AD, 32);

    const int n = lane & 15;
    const int r = nBlock * 16 + n;
    const int subj = rel_subj[r];
    const int obj  = rel_obj[r];
    const int mHi  = (lane >> 4) << 3;

#pragma unroll
    for (int mb = 0; mb < BSZ / 16; ++mb) {
        v16h a0 = frag_load(metaH, mb * 16, AD, 0);
        v16h a1 = frag_load(metaH, mb * 16, AD, 32);
        v8f acc = {};
        acc = wmma16(a0, b0, acc);
        acc = wmma16(a1, b1, acc);
#pragma unroll
        for (int e = 0; e < 8; ++e) {
            const int b = mb * 16 + mHi + e;
            float h = 1.0f / (1.0f + __expf(-acc[e]));
            float val = prev[(long)b * prevStride + subj] * h;
            atomicAdd(&nxt[(long)b * nxtStride + obj], val);
        }
    }
}

// ---------------------------------------------------------------------------
// Host-side orchestration
// ---------------------------------------------------------------------------
extern "C" void kernel_launch(void* const* d_in, const int* in_sizes, int n_in,
                              void* d_out, int out_size, void* d_ws, size_t ws_size,
                              hipStream_t stream) {
    const int*   x          = (const int*)  d_in[0];
    const int*   rel_subj   = (const int*)  d_in[1];
    const int*   rel_obj    = (const int*)  d_in[2];
    const float* action_emb = (const float*)d_in[3];
    const float* pos_emb    = (const float*)d_in[4];
    const float* state_emb  = (const float*)d_in[5];
    const float* s2r_w1     = (const float*)d_in[6];
    const float* s2r_b1     = (const float*)d_in[7];
    const float* s2r_w2     = (const float*)d_in[8];
    const float* s2r_b2     = (const float*)d_in[9];
    const float* q_w        = (const float*)d_in[10];
    const float* q_b        = (const float*)d_in[11];
    const float* ms_w1      = (const float*)d_in[12];
    const float* ms_b1      = (const float*)d_in[13];
    const float* ms_w2      = (const float*)d_in[14];
    const float* ms_b2      = (const float*)d_in[15];
    const float* meta_init  = (const float*)d_in[16];

    float* out = (float*)d_out;                // (B, L, SIZE)

    // workspace carve-up (256B aligned)
    char* ws = (char*)d_ws;
    size_t off = 0;
    auto carve = [&](size_t bytes) -> char* {
        char* p = ws + off;
        off = (off + bytes + 255) & ~(size_t)255;
        return p;
    };
    float*    initF    = (float*)   carve((size_t)BSZ * SIZE_ * 4);
    float*    sent     = (float*)   carve((size_t)BSZ * LEN * TWOA * 4);
    _Float16* sembH    = (_Float16*)carve((size_t)SIZE_ * SED_ * 2);
    _Float16* w1t      = (_Float16*)carve((size_t)HID_ * TWOA * 2);
    _Float16* w2t      = (_Float16*)carve((size_t)AD * HID_ * 2);
    _Float16* relH     = (_Float16*)carve((size_t)RR * AD * 2);
    float*    meta     = (float*)   carve((size_t)BSZ * AD * 4);
    _Float16* metaH    = (_Float16*)carve((size_t)BSZ * AD * 2);
    (void)ws_size; (void)n_in; (void)in_sizes; (void)out_size;

    const int outN = BSZ * LEN * SIZE_;
    k_zero<<<(outN + 255) / 256, 256, 0, stream>>>(out, outN);
    k_init_state<<<(BSZ * SIZE_ + 255) / 256, 256, 0, stream>>>(x, initF);
    k_sentence<<<(BSZ * LEN * TWOA + 255) / 256, 256, 0, stream>>>(
        x, action_emb, pos_emb, sent);
    k_f2h<<<(SIZE_ * SED_ + 255) / 256, 256, 0, stream>>>(state_emb, sembH,
                                                          SIZE_ * SED_);
    k_transpose_f2h<<<(TWOA * HID_ + 255) / 256, 256, 0, stream>>>(
        s2r_w1, w1t, TWOA, HID_);
    k_transpose_f2h<<<(HID_ * AD + 255) / 256, 256, 0, stream>>>(
        s2r_w2, w2t, HID_, AD);
    k_meta_init<<<(BSZ * AD + 255) / 256, 256, 0, stream>>>(meta_init, meta);

    // relation MLP: 4096 tiles of 16 rows, 8 waves/block
    relation_kernel<<<RR / 16 / 8, 256, 0, stream>>>(
        sembH, w1t, w2t, s2r_b1, s2r_b2, rel_subj, rel_obj, relH);

    // sequential scan: L steps
    for (int t = 0; t < LEN; ++t) {
        meta_step_kernel<<<BSZ, 256, 0, stream>>>(
            meta, metaH, sent, q_w, q_b, ms_w1, ms_b1, ms_w2, ms_b2);

        const float* prev = (t == 0) ? initF : (out + (size_t)(t - 1) * SIZE_);
        int prevStride = (t == 0) ? SIZE_ : (LEN * SIZE_);
        float* nxt = out + (size_t)t * SIZE_;

        // one wave per relation tile: R/16 = 4096 waves -> 512 blocks
        h_step_kernel<<<(RR / 16) / 8, 256, 0, stream>>>(
            metaH, relH, rel_subj, rel_obj, prev, prevStride, nxt, LEN * SIZE_);
    }
}
